// GraphLSTM_61607010894254
// MI455X (gfx1250) — matrix operand order repaired and checked by reading, not compiled, not run on hardware
//
#include <hip/hip_runtime.h>
#include <hip/hip_bf16.h>

// ---------------------------------------------------------------------------
// GraphLSTM on MI455X (gfx1250): f16 WMMA gates GEMM + fused LSTM pointwise.
// E=400000 edges, N=50000 nodes, B=50 graphs, D=64.
// ---------------------------------------------------------------------------

#define E_CNT 400000
#define N_CNT 50000
#define B_CNT 50
#define D_DIM 64

typedef _Float16 v16h __attribute__((ext_vector_type(16)));
typedef _Float16 v8h  __attribute__((ext_vector_type(8)));
typedef _Float16 v4h  __attribute__((ext_vector_type(4)));
typedef float    v8f  __attribute__((ext_vector_type(8)));

struct Segs {
    const float* base[5];  // each segment is a (rows x 64) f32 matrix
    const int*   idx[5];   // nullptr => identity row index
};

__device__ __forceinline__ float sig_(float x) { return 1.0f / (1.0f + __expf(-x)); }

// ---------------------------------------------------------------------------
// Pack [Wih | Whh] -> f16 row-major (256 x (kin+64)), bias = bih + bhh.
// ---------------------------------------------------------------------------
__global__ void pack_weights(const float* __restrict__ Wih, const float* __restrict__ Whh,
                             const float* __restrict__ bih, const float* __restrict__ bhh,
                             _Float16* __restrict__ W16, float* __restrict__ bias, int kin) {
    const int K = kin + D_DIM;
    const int total = 256 * K;
    for (int i = blockIdx.x * blockDim.x + threadIdx.x; i < total;
         i += gridDim.x * blockDim.x) {
        const int r = i / K, c = i % K;
        const float v = (c < kin) ? Wih[r * kin + c] : Whh[r * D_DIM + (c - kin)];
        W16[i] = (_Float16)v;
    }
    const int id = blockIdx.x * blockDim.x + threadIdx.x;
    if (id < 256) bias[id] = bih[id] + bhh[id];
}

__global__ void zero_f32(float* __restrict__ p, size_t n) {
    for (size_t i = blockIdx.x * (size_t)blockDim.x + threadIdx.x; i < n;
         i += (size_t)gridDim.x * blockDim.x) p[i] = 0.0f;
}

// ---------------------------------------------------------------------------
// Fused gather -> f16 WMMA gates GEMM -> LSTM cell -> stores + atomic scatter.
// Block: 128 threads = 4 wave32. 32 rows per block, 256 gate columns.
// Wave w: M-tile (w>>1), column-pairs t in {2*(w&1), 2*(w&1)+1}; for each t it
// accumulates gate tiles i,f,g,o of the SAME 16 columns so the cell update is
// done entirely in accumulator registers.
// ---------------------------------------------------------------------------
template <int NSEG>
__global__ void lstm_wmma_kernel(Segs segs,
                                 const _Float16* __restrict__ W16,
                                 const float* __restrict__ bias,
                                 const float* __restrict__ c_in,
                                 float* __restrict__ h_out,
                                 float* __restrict__ c_out,
                                 float* __restrict__ relu_out,
                                 float* acc0, const int* acc0_idx,
                                 float* acc1, const int* acc1_idx,
                                 int nrows) {
    constexpr int K  = NSEG * D_DIM;   // 320 (edge) or 256 (node/graph)
    constexpr int KT = K / 32;         // k-steps of 32

    __shared__ __align__(16) _Float16 As[32 * K];  // 32-row f16 A tile

    const int tid  = threadIdx.x;
    const int row0 = blockIdx.x * 32;

    // ---- stage x-rows (gather + f32->f16) into LDS -------------------------
#pragma unroll
    for (int s = 0; s < NSEG; ++s) {
        const float* __restrict__ base = segs.base[s];
        const int*   __restrict__ idx  = segs.idx[s];
        for (int i = tid; i < 32 * 16; i += 128) {      // 16 float4 per row
            const int row  = i >> 4;
            const int c4   = (i & 15) * 4;
            int grow = row0 + row;
            grow = (grow < nrows) ? grow : (nrows - 1);  // clamp pad rows
            const int srow = idx ? idx[grow] : grow;
            const float4 v = *(const float4*)(base + (size_t)srow * D_DIM + c4);
            v4h hv;
            hv[0] = (_Float16)v.x; hv[1] = (_Float16)v.y;
            hv[2] = (_Float16)v.z; hv[3] = (_Float16)v.w;
            *(v4h*)&As[row * K + s * D_DIM + c4] = hv;
        }
    }
    __syncthreads();

    // ---- per-wave WMMA ----------------------------------------------------
    const int w     = tid >> 5;        // wave id (wave32)
    const int lane  = tid & 31;
    const int mtile = w >> 1;          // 0..1  (16 rows each)
    const int arow  = lane & 15;       // A: row within tile
    const int kh    = lane >> 4;       // A/B: k-half selector
    const int bcol  = lane & 15;       // B: output column within tile
    const int bk    = kh * 16;         // B: k offset (lanes 16-31 -> K 16..31)

#pragma unroll
    for (int t2 = 0; t2 < 2; ++t2) {
        const int t = (w & 1) * 2 + t2;       // 16-col slice within each gate
        v8f acc[4] = {v8f{}, v8f{}, v8f{}, v8f{}};   // gates i,f,g,o

        for (int kt = 0; kt < KT; ++kt) {
            // A operand: contiguous k order (lane<16: K0..15; else 16..31), matching B
            const _Float16* ap = &As[(mtile * 16 + arow) * K + kt * 32 + kh * 16];
            const v8h a0 = *(const v8h*)ap;
            const v8h a1 = *(const v8h*)(ap + 8);
            const v16h a = __builtin_shufflevector(a0, a1,
                0, 1, 2, 3, 4, 5, 6, 7, 8, 9, 10, 11, 12, 13, 14, 15);
#pragma unroll
            for (int g = 0; g < 4; ++g) {
                // B(k,n) = Wcat[n][k]; column n = W row (g*64 + t*16 + bcol)
                const int wrow = g * D_DIM + t * 16 + bcol;
                const v16h b = *(const v16h*)(W16 + (size_t)wrow * K + kt * 32 + bk);
                acc[g] = __builtin_amdgcn_wmma_f32_16x16x32_f16(
                    false, a, false, b, (short)0, acc[g], false, false);
            }
        }

        // ---- fused LSTM cell straight out of accumulators -------------------
        const int col = t * 16 + (lane & 15);        // 0..63 within D
#pragma unroll
        for (int g = 0; g < 4; ++g) {
            const float bi = bias[g * D_DIM + col];
#pragma unroll
            for (int r = 0; r < 8; ++r) acc[g][r] += bi;
        }
#pragma unroll
        for (int r = 0; r < 8; ++r) {
            const int grow = row0 + mtile * 16 + r + kh * 8;  // C/D layout row
            if (grow < nrows) {
                const size_t o = (size_t)grow * D_DIM + col;
                const float iv = sig_(acc[0][r]);
                const float fv = sig_(acc[1][r]);
                const float gv = tanhf(acc[2][r]);
                const float ov = sig_(acc[3][r]);
                const float c2 = fv * c_in[o] + iv * gv;
                const float hh = ov * tanhf(c2);
                const float re = hh > 0.0f ? hh : 0.0f;
                h_out[o]    = hh;
                c_out[o]    = c2;
                relu_out[o] = re;
                if (acc0) atomicAdd(acc0 + (size_t)acc0_idx[grow] * D_DIM + col, re);
                if (acc1) atomicAdd(acc1 + (size_t)acc1_idx[grow] * D_DIM + col, re);
            }
        }
    }
}

// ---------------------------------------------------------------------------
// Host launcher
// ---------------------------------------------------------------------------
extern "C" void kernel_launch(void* const* d_in, const int* in_sizes, int n_in,
                              void* d_out, int out_size, void* d_ws, size_t ws_size,
                              hipStream_t stream) {
    const float* edge_feat = (const float*)d_in[0];
    const float* node_feat = (const float*)d_in[1];
    const float* g_repr    = (const float*)d_in[2];
    const float* edge_h    = (const float*)d_in[3];
    const float* edge_c    = (const float*)d_in[4];
    const float* node_h    = (const float*)d_in[5];
    const float* node_c    = (const float*)d_in[6];
    const float* graph_h   = (const float*)d_in[7];
    const float* graph_c   = (const float*)d_in[8];
    const float* Wih_e = (const float*)d_in[9];
    const float* Whh_e = (const float*)d_in[10];
    const float* bih_e = (const float*)d_in[11];
    const float* bhh_e = (const float*)d_in[12];
    const float* Wih_n = (const float*)d_in[13];
    const float* Whh_n = (const float*)d_in[14];
    const float* bih_n = (const float*)d_in[15];
    const float* bhh_n = (const float*)d_in[16];
    const float* Wih_u = (const float*)d_in[17];
    const float* Whh_u = (const float*)d_in[18];
    const float* bih_u = (const float*)d_in[19];
    const float* bhh_u = (const float*)d_in[20];
    const int* src        = (const int*)d_in[21];
    const int* dst        = (const int*)d_in[22];
    const int* edge_graph = (const int*)d_in[23];
    const int* node_graph = (const int*)d_in[24];

    float* out = (float*)d_out;
    const size_t E64 = (size_t)E_CNT * D_DIM;
    const size_t N64 = (size_t)N_CNT * D_DIM;
    const size_t B64 = (size_t)B_CNT * D_DIM;
    float* e_out = out;
    float* h_e   = out + E64;
    float* c_e   = out + 2 * E64;
    float* n_out = out + 3 * E64;
    float* h_n   = out + 3 * E64 + N64;
    float* c_n   = out + 3 * E64 + 2 * N64;
    float* u_out = out + 3 * E64 + 3 * N64;
    float* h_u   = u_out + B64;
    float* c_u   = u_out + 2 * B64;

    // workspace layout (256B aligned)
    char* ws = (char*)d_ws;
    _Float16* W16e  = (_Float16*)(ws + 0);                 // 256*320*2 = 163840
    _Float16* W16n  = (_Float16*)(ws + 163840);            // 256*256*2 = 131072
    _Float16* W16u  = (_Float16*)(ws + 294912);            // 131072
    float*    biase = (float*)(ws + 425984);               // 1024
    float*    biasn = (float*)(ws + 427008);
    float*    biasu = (float*)(ws + 428032);
    float*    mail  = (float*)(ws + 429056);               // N*64*4 = 12.8 MB
    float*    ecomb = (float*)(ws + 429056 + N64 * 4);     // B*64*4
    float*    ncomb = ecomb + B64;                         // B*64*4 (contiguous)

    // 1) pack weights to f16 + fold biases
    pack_weights<<<320, 256, 0, stream>>>(Wih_e, Whh_e, bih_e, bhh_e, W16e, biase, 256);
    pack_weights<<<256, 256, 0, stream>>>(Wih_n, Whh_n, bih_n, bhh_n, W16n, biasn, 192);
    pack_weights<<<256, 256, 0, stream>>>(Wih_u, Whh_u, bih_u, bhh_u, W16u, biasu, 192);

    // 2) zero scatter accumulators (graph-replay deterministic structure)
    zero_f32<<<12500, 256, 0, stream>>>(mail, N64);
    zero_f32<<<13, 256, 0, stream>>>(ecomb, 2 * B64);  // ecomb + ncomb contiguous

    // 3) edge cell: x = [edge_feat | nf[src] | nf[dst] | g_repr[eg] | edge_h]
    {
        Segs s{};
        s.base[0] = edge_feat; s.idx[0] = nullptr;
        s.base[1] = node_feat; s.idx[1] = src;
        s.base[2] = node_feat; s.idx[2] = dst;
        s.base[3] = g_repr;    s.idx[3] = edge_graph;
        s.base[4] = edge_h;    s.idx[4] = nullptr;
        lstm_wmma_kernel<5><<<(E_CNT + 31) / 32, 128, 0, stream>>>(
            s, W16e, biase, edge_c, h_e, c_e, e_out,
            mail, dst, ecomb, edge_graph, E_CNT);
    }

    // 4) node cell: x = [node_feat | mail | g_repr[ng] | node_h]
    {
        Segs s{};
        s.base[0] = node_feat; s.idx[0] = nullptr;
        s.base[1] = mail;      s.idx[1] = nullptr;
        s.base[2] = g_repr;    s.idx[2] = node_graph;
        s.base[3] = node_h;    s.idx[3] = nullptr;
        lstm_wmma_kernel<4><<<(N_CNT + 31) / 32, 128, 0, stream>>>(
            s, W16n, biasn, node_c, h_n, c_n, n_out,
            ncomb, node_graph, nullptr, nullptr, N_CNT);
    }

    // 5) graph cell: x = [n_comb | e_comb | g_repr | graph_h]
    {
        Segs s{};
        s.base[0] = ncomb;   s.idx[0] = nullptr;
        s.base[1] = ecomb;   s.idx[1] = nullptr;
        s.base[2] = g_repr;  s.idx[2] = nullptr;
        s.base[3] = graph_h; s.idx[3] = nullptr;
        lstm_wmma_kernel<4><<<(B_CNT + 31) / 32, 128, 0, stream>>>(
            s, W16u, biasu, graph_c, h_u, c_u, u_out,
            nullptr, nullptr, nullptr, nullptr, B_CNT);
    }
}
